// QRNNLayer_14370960573041
// MI455X (gfx1250) — compile-verified, matrix operand order checked by reading, not run on hardware
//
#include <hip/hip_runtime.h>
#include <hip/hip_bf16.h>

// ---------------- problem constants (match reference) ----------------
#define LT    2048        // sequence length
#define BB    16          // batch
#define DIN   512         // input dim
#define DOUT  512         // output dim per direction
#define NG    1536        // 3*DOUT gate columns
#define MROWS (LT*BB)     // 32768 GEMM rows

typedef __attribute__((ext_vector_type(16))) __bf16 v16bf;
typedef __attribute__((ext_vector_type(8)))  __bf16 v8bf;
typedef __attribute__((ext_vector_type(8)))  float  v8f;
typedef __attribute__((ext_vector_type(4)))  unsigned int v4u;
typedef __attribute__((ext_vector_type(8)))  int    v8i;
typedef __attribute__((ext_vector_type(4)))  int    v4i;

#if defined(__HIP_DEVICE_COMPILE__) && defined(__has_builtin)
# if __has_builtin(__builtin_amdgcn_tensor_load_to_lds) && __has_builtin(__builtin_amdgcn_s_wait_tensorcnt)
#  define USE_TDM 1
# endif
#endif

// ---------------- conversion kernels ----------------
__global__ void cvt_x_bf16(const float* __restrict__ x, __bf16* __restrict__ xb) {
    int i = blockIdx.x * blockDim.x + threadIdx.x;   // covers MROWS*DIN = 16,777,216
    xb[i] = (__bf16)x[i];
}

// W: [3][DIN][NG] f32  ->  Wt: [3][NG][DIN] bf16   (column-major per tap, so B
// fragments are two contiguous 16B chunks per lane, same layout as A fragments)
__global__ void cvt_w_bf16(const float* __restrict__ W, __bf16* __restrict__ Wt) {
    int i = blockIdx.x * blockDim.x + threadIdx.x;   // covers 3*NG*DIN = 2,359,296
    int tap = i / (NG * DIN);
    int r   = i % (NG * DIN);
    int n   = r / DIN;
    int k   = r % DIN;
    Wt[i] = (__bf16)W[(size_t)tap * DIN * NG + (size_t)k * NG + n];
}

#ifdef USE_TDM
// Issue one TDM descriptor: load a [Z=3 taps][Y=64 cols][X=32 K] bf16 tile of
// W^T ([3][NG][DIN], row stride DIN, tap stride NG*DIN) into LDS, contiguous as
// [tap][col][32K].  Built per CDNA5 ISA ch.8 D# layout (groups 0..3; the
// toolchain's 6-arg builtin takes an extra int32x8 block, passed as zeros).
__device__ __forceinline__ void tdm_load_B(const __bf16* gsrc, unsigned lds_off) {
    unsigned long long ga = (unsigned long long)(uintptr_t)gsrc;  // 57-bit byte addr
    v4u g0 = {
        1u,                                                   // count=1, no gather
        lds_off,                                              // lds_addr[31:0]
        (unsigned)ga,                                         // global_addr[31:0]
        ((unsigned)(ga >> 32) & 0x01FFFFFFu) | 0x80000000u    // addr[56:32] | type=2
    };
    v8i g1 = {
        (int)0x00010000u,          // wg_mask=0 (not in cluster), data_size=1 (2B)
        (int)(512u  << 16),        // tensor_dim0[15:0]=512 (row length, elems)
        (int)(1536u << 16),        // tensor_dim0 hi=0 | tensor_dim1[15:0]=1536 rows
        (int)(32u   << 16),        // tensor_dim1 hi=0 | tile_dim0=32 (K chunk)
        (int)((3u << 16) | 64u),   // tile_dim1=64 cols | tile_dim2=3 taps
        512,                       // tensor_dim0_stride[31:0] = DIN
        0,                         // dim0_stride hi | dim1_stride[15:0] (786432&0xFFFF=0)
        12                         // tensor_dim1_stride[47:16] = (NG*DIN)>>16
    };
    v4i g2 = { 3, 0, 0, 0 };       // tensor_dim2=3; no dim3/iterate
    v4i g3 = { 0, 0, 0, 0 };
    v8i g4 = { 0, 0, 0, 0, 0, 0, 0, 0 };   // extension block (unused) -> zeros
    __builtin_amdgcn_tensor_load_to_lds(g0, g1, g2, g3, g4, 0);
}
#endif

// ---------------- gate GEMM ----------------
// Block = 4 waves = 4 consecutive timesteps, sharing one 64-column weight slab.
// Wave w computes gates[t0+w, :, n0..n0+63]: 16(M=batch) x 64(N) x 512(K) x 3 taps.
// Weights stream through a 2 x 12KB LDS double buffer filled by the TDM.
__global__ __launch_bounds__(128)
void gemm_gates(const __bf16* __restrict__ xb,     // [MROWS][DIN] bf16
                const __bf16* __restrict__ Wt,     // [3][NG][DIN] bf16
                const float*  __restrict__ bias,   // [3][NG] f32
                float* __restrict__ gates) {       // [MROWS][NG] f32 (activated)
    __shared__ __bf16 ldsB[2][3 * 64 * 32];        // [buf][tap][col][K-chunk]

    const int lane  = threadIdx.x & 31;
    // wave index is physically uniform across the wave32: tell the compiler so
    // it uses scalar branches (not saveexec) and SALU address bases.
    const int wave  = __builtin_amdgcn_readfirstlane(threadIdx.x >> 5);
    const int ng    = blockIdx.x % 24;
    const int t     = (blockIdx.x / 24) * 4 + wave;          // uniform per wave
    const int n0    = ng * 64;
    const int mlane = lane & 15;                   // row (A) / col (B) within tile
    const int khalf = (lane >> 4) * 8;             // lane-group K offset; also C-row offset

    v8f acc[4];
    const v8f zero = {0.f,0.f,0.f,0.f,0.f,0.f,0.f,0.f};
    #pragma unroll
    for (int i = 0; i < 4; ++i) acc[i] = zero;

    const __bf16* Bg = Wt + (size_t)n0 * DIN;      // weight tile base (tap 0, col n0, k 0)

#ifdef USE_TDM
    // flat __shared__ pointers carry the LDS byte offset in their low 32 bits
    // (ISA 10.2 aperture mapping)
    const unsigned ldsOff[2] = { (unsigned)(uintptr_t)&ldsB[0][0],
                                 (unsigned)(uintptr_t)&ldsB[1][0] };
    if (wave == 0) {
        tdm_load_B(Bg, ldsOff[0]);                         // prologue: chunk 0
        __builtin_amdgcn_s_wait_tensorcnt((short)0);
    }
    __syncthreads();
#endif

    for (int ki = 0; ki < DIN / 32; ++ki) {
        const int kb = ki * 32;
#ifdef USE_TDM
        if (wave == 0 && ki + 1 < DIN / 32)                // prefetch next chunk
            tdm_load_B(Bg + kb + 32, ldsOff[(ki + 1) & 1]);
        const __bf16* Bl = &ldsB[ki & 1][0];
#else
        __syncthreads();                                   // prev reads done
        for (int i = threadIdx.x; i < 768; i += 128) {     // 768 x 16B = 12KB chunk
            int r = i >> 2, c = i & 3;                     // r: tap*64+col, c: 16B sub
            int tap = r >> 6, col = r & 63;
            *(v8bf*)&ldsB[0][r * 32 + c * 8] =
                *(const v8bf*)(Wt + ((size_t)tap * NG + n0 + col) * DIN + kb + c * 8);
        }
        __syncthreads();
        const __bf16* Bl = &ldsB[0][0];
#endif
        #pragma unroll
        for (int tap = 0; tap < 3; ++tap) {
            const int tt = t - 1 + tap;                    // uniform tap guard
            if (tt < 0 || tt >= LT) continue;
            const __bf16* A = xb + ((size_t)tt * BB + mlane) * DIN + kb + khalf;
            v8bf alo = *(const v8bf*)(A);
            v8bf ahi = *(const v8bf*)(A + 16);
            v16bf af = __builtin_shufflevector(alo, ahi,
                         0,1,2,3,4,5,6,7,8,9,10,11,12,13,14,15);
            #pragma unroll
            for (int nt = 0; nt < 4; ++nt) {
                const __bf16* Bc = Bl + ((tap * 64 + nt * 16 + mlane) * 32 + khalf);
                v8bf blo = *(const v8bf*)(Bc);
                v8bf bhi = *(const v8bf*)(Bc + 16);
                v16bf bf_ = __builtin_shufflevector(blo, bhi,
                             0,1,2,3,4,5,6,7,8,9,10,11,12,13,14,15);
                acc[nt] = __builtin_amdgcn_wmma_f32_16x16x32_bf16(
                              false, af, false, bf_, (short)0, acc[nt], false, false);
            }
        }
#ifdef USE_TDM
        if (wave == 0) __builtin_amdgcn_s_wait_tensorcnt((short)0);
        __syncthreads();                                   // publish next buffer
#endif
    }

    // epilogue: boundary-aware bias + sigmoid/tanh, store f32 gates
    const float* b0 = bias;             // tap 0 bias: only when t > 0
    const float* b1 = bias + NG;        // center tap: always
    const float* b2 = bias + 2 * NG;    // tap 2 bias: only when t < L-1
    #pragma unroll
    for (int nt = 0; nt < 4; ++nt) {
        const int col = n0 + nt * 16 + mlane;
        float bs = b1[col];
        if (t > 0)      bs += b0[col];
        if (t < LT - 1) bs += b2[col];
        const bool isTanh = (col >= 2 * DOUT);     // z-gate region
        #pragma unroll
        for (int j = 0; j < 8; ++j) {
            const int m = j + khalf;               // C layout: VGPR j, lanes>=16 -> M=j+8
            float v = acc[nt][j] + bs;
            float r = isTanh ? tanhf(v) : (1.f / (1.f + __expf(-v)));
            gates[((size_t)t * BB + m) * NG + col] = r;
        }
    }
}

// ---------------- fo-pool scan: one thread per (batch, channel) ----------------
__global__ void fo_pool(const float* __restrict__ gates,  // [MROWS][NG] activated
                        float* __restrict__ out,          // [L][B][1024]
                        float* __restrict__ lastH,        // [B][1024]
                        float* __restrict__ lastC,        // [B][1024]
                        int dir) {                        // 0=fwd, 1=rev
    const int tid = blockIdx.x * blockDim.x + threadIdx.x;  // 0..8191
    const int b   = tid >> 9;        // /512
    const int ch  = tid & 511;
    float c = 0.f;
    for (int s = 0; s < LT; ++s) {
        const int t = dir ? (LT - 1 - s) : s;
        const float* g = gates + ((size_t)t * BB + b) * NG + ch;
        const float f = g[0];
        const float o = g[DOUT];
        const float z = g[2 * DOUT];
        c = f * z + (1.f - f) * c;
        const float h = c * o;
        out[((size_t)t * BB + b) * (2 * DOUT) + dir * DOUT + ch] = h;
        // reference takes index [-1] (t == L-1) for BOTH directions:
        // fwd -> final scan step; rev -> its FIRST scan step.
        if (t == LT - 1) {
            lastH[b * (2 * DOUT) + dir * DOUT + ch] = h;
            lastC[b * (2 * DOUT) + dir * DOUT + ch] = c;
        }
    }
}

// ---------------- launcher ----------------
extern "C" void kernel_launch(void* const* d_in, const int* in_sizes, int n_in,
                              void* d_out, int out_size, void* d_ws, size_t ws_size,
                              hipStream_t stream) {
    (void)in_sizes; (void)n_in; (void)out_size; (void)ws_size;
    const float* input = (const float*)d_in[0];
    // d_in[1] = lengths: unused by the reference forward body
    const float* fwd_W = (const float*)d_in[2];
    const float* fwd_b = (const float*)d_in[3];
    const float* rev_W = (const float*)d_in[4];
    const float* rev_b = (const float*)d_in[5];

    // workspace layout (all offsets 256B-aligned); gate buffer reused per direction
    char* ws = (char*)d_ws;
    const size_t xb_bytes = (size_t)MROWS * DIN * sizeof(__bf16);     // 33,554,432
    const size_t wt_bytes = (size_t)3 * NG * DIN * sizeof(__bf16);    //  4,718,592
    __bf16* xb    = (__bf16*)ws;
    __bf16* WtF   = (__bf16*)(ws + xb_bytes);
    __bf16* WtR   = (__bf16*)(ws + xb_bytes + wt_bytes);
    float*  gates = (float*) (ws + xb_bytes + 2 * wt_bytes);          // 201,326,592

    float* out   = (float*)d_out;                       // [L][B][1024]
    float* lastH = out + (size_t)LT * BB * 2 * DOUT;    // [B][1024]
    float* lastC = lastH + (size_t)BB * 2 * DOUT;       // [B][1024]

    // 1) precision conversion (bf16 activations + transposed bf16 weights; L2-resident)
    cvt_x_bf16<<<(MROWS * DIN) / 256, 256, 0, stream>>>(input, xb);
    cvt_w_bf16<<<(3 * NG * DIN) / 256, 256, 0, stream>>>(fwd_W, WtF);
    cvt_w_bf16<<<(3 * NG * DIN) / 256, 256, 0, stream>>>(rev_W, WtR);

    // 2) forward direction: TDM-fed WMMA gate GEMM (fused bias+activation) then scan
    const int gemm_blocks = (LT / 4) * 24;   // 4 waves/block, one timestep/wave
    gemm_gates<<<gemm_blocks, 128, 0, stream>>>(xb, WtF, fwd_b, gates);
    fo_pool<<<32, 256, 0, stream>>>(gates, out, lastH, lastC, 0);

    // 3) reverse direction (reuses the gate buffer; stream order guarantees safety)
    gemm_gates<<<gemm_blocks, 128, 0, stream>>>(xb, WtR, rev_b, gates);
    fo_pool<<<32, 256, 0, stream>>>(gates, out, lastH, lastC, 1);
}
